// VQVAE_58866821759618
// MI455X (gfx1250) — compile-verified, matrix operand order checked by reading, not run on hardware
//
#include <hip/hip_runtime.h>
#include <hip/hip_bf16.h>

typedef __attribute__((ext_vector_type(16))) _Float16 v16h;
typedef __attribute__((ext_vector_type(8)))  float    v8f;
typedef __attribute__((ext_vector_type(4)))  unsigned int u32x4;
typedef __attribute__((ext_vector_type(8)))  int      i32x8;
typedef __attribute__((ext_vector_type(4)))  int      i32x4;

#define NROWS    262144
#define D_X      64
#define D_Z      32
#define NBINS    1024
#define GAMMA_F  0.99f
#define ONE_MG   0.01f
#define BETA_F   0.25f
#define MIN_ASSIGN_F (0.1f / 1024.0f)
#define LOG_2PI  1.8378770664093453f

#if __has_builtin(__builtin_amdgcn_tensor_load_to_lds) && \
    __has_builtin(__builtin_amdgcn_s_wait_tensorcnt)
#define USE_TDM 1
#else
#define USE_TDM 0
#endif

// CDNA5 16-bit A-matrix lane/k mapping (ISA 7.12.2): element e (0..15) of the
// per-lane v16h, half h = lane>>4.  VGPR v=e/2, pair pos p=e&1:
//   v<4 : k = 2v+p + 8h      (K 0..7  / 8..15)
//   v>=4: k = 16+2(v-4)+p+8h (K 16..23 / 24..31)
__device__ __forceinline__ int amap_k(int e, int h) {
    int v = e >> 1, p = e & 1;
    int k = (v < 4) ? (2 * v + p) : (16 + 2 * (v - 4) + p);
    return k + 8 * h;
}

__device__ __forceinline__ v8f wmma_f16(v16h a, v16h b, v8f c) {
    return __builtin_amdgcn_wmma_f32_16x16x32_f16(false, a, false, b, (short)0, c,
                                                  false, false);
}

// ---------------------------------------------------------------------------
// Prep: zero accumulators, decoded[b][j] = (v_b @ W_dec + b_dec),
// vnorm[b] = ||v_b||^2, and pack vectors / W_enc into WMMA B-operand layout.
// ---------------------------------------------------------------------------
__global__ __launch_bounds__(256) void vq_prep(
    const float* __restrict__ W_enc, const float* __restrict__ vectors,
    const float* __restrict__ W_dec, const float* __restrict__ b_dec,
    float* __restrict__ decoded, float* __restrict__ vnorm,
    float* __restrict__ counts, float* __restrict__ sumz,
    float* __restrict__ lossAcc, _Float16* __restrict__ vecsB,
    _Float16* __restrict__ wencB)
{
    int tid    = blockIdx.x * blockDim.x + threadIdx.x;
    int stride = gridDim.x * blockDim.x;

    for (int q = tid; q < NBINS; q += stride) counts[q] = 0.0f;
    for (int q = tid; q < NBINS * D_Z; q += stride) sumz[q] = 0.0f;
    if (tid < 2) lossAcc[tid] = 0.0f;

    for (int q = tid; q < NBINS; q += stride) {
        float s = 0.0f;
        for (int k = 0; k < D_Z; ++k) { float v = vectors[q * D_Z + k]; s += v * v; }
        vnorm[q] = s;
    }
    for (int q = tid; q < NBINS * D_X; q += stride) {
        int b = q >> 6, j = q & 63;
        float s = b_dec[j];
        for (int k = 0; k < D_Z; ++k) s += vectors[b * D_Z + k] * W_dec[k * D_X + j];
        decoded[q] = s;
    }
    // vecsB[t][lane][e] : B tile for bins t*16..t*16+15 (64 tiles)
    for (int q = tid; q < 64 * 32 * 16; q += stride) {
        int e = q & 15, lane = (q >> 4) & 31, t = q >> 9;
        vecsB[q] = (_Float16)vectors[(t * 16 + (lane & 15)) * D_Z + amap_k(e, lane >> 4)];
    }
    // wencB[kc][nt][lane][e] : W_enc B tiles, K chunks kc=0,1 ; N tiles nt=0,1
    for (int q = tid; q < 2 * 2 * 32 * 16; q += stride) {
        int e = q & 15, lane = (q >> 4) & 31, nt = (q >> 9) & 1, kc = q >> 10;
        wencB[q] = (_Float16)W_enc[(kc * 32 + amap_k(e, lane >> 4)) * D_Z +
                                   (nt * 16 + (lane & 15))];
    }
}

// ---------------------------------------------------------------------------
// Main fused kernel: 128 threads = 4 wave32, 32 rows per wave (M-blocked x2 so
// each B tile feeds two WMMAs), 128 rows/block.
// z = x@W_enc+b_enc (WMMA); argmin_b(||v_b||^2 - 2 z.v_b) over 64 WMMA tiles;
// recon/commit loss partials; counts/sum_z histograms.
// x tile staged into LDS via the Tensor Data Mover when available.
// ---------------------------------------------------------------------------
__global__ __launch_bounds__(128) void vq_main(
    const float* __restrict__ x, const float* __restrict__ b_enc,
    const float* __restrict__ vectors, const float* __restrict__ decoded,
    const float* __restrict__ vnorm, const _Float16* __restrict__ vecsB,
    const _Float16* __restrict__ wencB, float* __restrict__ counts,
    float* __restrict__ sumz, float* __restrict__ lossAcc)
{
    __shared__ float sX[128 * D_X];   // 32 KB
    __shared__ float sZ[128 * D_Z];   // 16 KB
    __shared__ int   sIdx[128];
    __shared__ float sLoss[2];

    const int tid  = threadIdx.x;
    const int w    = tid >> 5;
    const int lane = tid & 31;
    const int m    = lane & 15;       // row (A) / col (B,C) position
    const int h    = lane >> 4;       // half-wave

    if (tid == 0) { sLoss[0] = 0.0f; sLoss[1] = 0.0f; }

    // ---- stage x tile (128 x 64 f32 = 32KB) into LDS ----
    const unsigned long long gaddr =
        (unsigned long long)(const void*)(x + (size_t)blockIdx.x * 128 * D_X);
#if USE_TDM
    if (w == 0) {
        // Tensor DMA descriptor (ISA 08 section 8.3/8.4), 2D tile:
        // data_size=4B, tensor/tile dim0 = 64, dim1 = 128, dim0_stride = 64.
        unsigned int ldsA = (unsigned int)(unsigned long long)(void*)sX;
        u32x4 g0;
        g0[0] = 1u;                                        // count=1, user desc
        g0[1] = ldsA;                                      // lds_addr
        g0[2] = (unsigned int)(gaddr & 0xFFFFFFFFull);     // global_addr lo
        g0[3] = (unsigned int)((gaddr >> 32) & 0x01FFFFFFull) | (2u << 30); // hi | type=2
        i32x8 g1;
        g1[0] = (int)0x00020000;        // wg_mask=0, data_size=2 (4B)
        g1[1] = (int)(64u << 16);       // tensor_dim0[15:0]=64 at bits 63:48
        g1[2] = (int)(128u << 16);      // tensor_dim0 hi=0 | tensor_dim1 lo=128
        g1[3] = (int)(64u << 16);       // tensor_dim1 hi=0 | tile_dim0=64
        g1[4] = (int)128;               // tile_dim1=128, tile_dim2=0
        g1[5] = (int)64;                // tensor_dim0_stride=64
        g1[6] = 0;                      // stride hi | tensor_dim1_stride lo
        g1[7] = 0;
        i32x4 gz4 = {0, 0, 0, 0};
        i32x8 gz8 = {0, 0, 0, 0, 0, 0, 0, 0};
        // 6-arg form (clang-23 / therock-10.0 headers)
        __builtin_amdgcn_tensor_load_to_lds(g0, g1, gz4, gz4, gz8, 0);
        __builtin_amdgcn_s_wait_tensorcnt(0);
    }
#else
    {
        const float4* xg = (const float4*)(const void*)(size_t)gaddr;
        float4* sx4 = (float4*)sX;
        for (int i = tid; i < 128 * D_X / 4; i += 128) sx4[i] = xg[i];
    }
#endif
    __builtin_prefetch(vecsB + (size_t)lane * 16, 0, 0);
    __syncthreads();

    const int wrow = w * 32;          // this wave's row base within block

    // ---- z = x @ W_enc + b_enc : 2 M-tiles x 2 K-chunks x 2 N-tiles WMMAs ----
    const float be0 = b_enc[m], be1 = b_enc[16 + m];
#pragma unroll
    for (int mt = 0; mt < 2; ++mt) {
        const int rbase = wrow + mt * 16;
        v8f c0 = {}; v8f c1 = {};
#pragma unroll
        for (int kc = 0; kc < 2; ++kc) {
            v16h a;
#pragma unroll
            for (int e = 0; e < 16; ++e)
                a[e] = (_Float16)sX[(rbase + m) * D_X + kc * 32 + amap_k(e, h)];
            v16h b0 = *(const v16h*)(wencB + ((size_t)(kc * 2 + 0) * 32 + lane) * 16);
            c0 = wmma_f16(a, b0, c0);
            v16h b1 = *(const v16h*)(wencB + ((size_t)(kc * 2 + 1) * 32 + lane) * 16);
            c1 = wmma_f16(a, b1, c1);
        }
#pragma unroll
        for (int r = 0; r < 8; ++r) { c0[r] += be0; c1[r] += be1; }
        // C layout: VGPR r holds row M=r+8h, column N=m  ->  spill z to LDS
#pragma unroll
        for (int r = 0; r < 8; ++r) {
            sZ[(rbase + r + 8 * h) * D_Z + m]      = c0[r];
            sZ[(rbase + r + 8 * h) * D_Z + 16 + m] = c1[r];
        }
    }
    __syncthreads();

    // ---- A-operands for the distance GEMM: two 16-row z tiles (f16) ----
    v16h az[2];
#pragma unroll
    for (int mt = 0; mt < 2; ++mt)
#pragma unroll
        for (int e = 0; e < 16; ++e)
            az[mt][e] = (_Float16)sZ[(wrow + mt * 16 + m) * D_Z + amap_k(e, h)];

    // ---- sweep all 1024 bins: 64 B tiles, 2 WMMAs per tile (M-blocked) ----
    float bestV[2][8]; int bestI[2][8];
#pragma unroll
    for (int mt = 0; mt < 2; ++mt)
#pragma unroll
        for (int r = 0; r < 8; ++r) { bestV[mt][r] = 3.4e38f; bestI[mt][r] = 0; }

#pragma unroll 4
    for (int t = 0; t < 64; ++t) {
        v16h b = *(const v16h*)(vecsB + ((size_t)t * 32 + lane) * 16);
        __builtin_prefetch(vecsB + ((size_t)(t + 8) * 32 + lane) * 16, 0, 0);
        v8f zero = {};
        v8f s0 = wmma_f16(az[0], b, zero);
        v8f s1 = wmma_f16(az[1], b, zero);
        float vn = vnorm[t * 16 + m];
        int   bi = t * 16 + m;
#pragma unroll
        for (int r = 0; r < 8; ++r) {
            float v0 = vn - 2.0f * s0[r];
            bool b0 = v0 < bestV[0][r];
            bestV[0][r] = b0 ? v0 : bestV[0][r];
            bestI[0][r] = b0 ? bi : bestI[0][r];
            float v1 = vn - 2.0f * s1[r];
            bool b1 = v1 < bestV[1][r];
            bestV[1][r] = b1 ? v1 : bestV[1][r];
            bestI[1][r] = b1 ? bi : bestI[1][r];
        }
    }

    // ---- reduce argmin across the 16 lanes of each half (xor<=8 stays in half)
#pragma unroll
    for (int mt = 0; mt < 2; ++mt) {
#pragma unroll
        for (int r = 0; r < 8; ++r) {
#pragma unroll
            for (int off = 1; off < 16; off <<= 1) {
                float ov = __shfl_xor(bestV[mt][r], off);
                int   oi = __shfl_xor(bestI[mt][r], off);
                if (ov < bestV[mt][r] ||
                    (ov == bestV[mt][r] && oi < bestI[mt][r])) {
                    bestV[mt][r] = ov; bestI[mt][r] = oi;
                }
            }
        }
        if (m == 0) {                 // lanes 0 and 16 publish their 8 rows
#pragma unroll
            for (int r = 0; r < 8; ++r)
                sIdx[wrow + mt * 16 + r + 8 * h] = bestI[mt][r];
        }
    }
    __syncthreads();

    // ---- per-row tail: 1 lane per row (32 rows per wave) ----
    const int row = wrow + lane;
    const int idx = sIdx[row];

    float commit = 0.0f, recon = 0.0f;
    for (int k = 0; k < D_Z; ++k) {
        float zv = sZ[row * D_Z + k];
        float d  = vectors[idx * D_Z + k] - zv;
        commit += d * d;
        atomicAdd(&sumz[idx * D_Z + k], zv);
    }
    for (int j = 0; j < D_X; ++j) {
        float d = sX[row * D_X + j] - decoded[idx * D_X + j];
        recon += d * d;
    }
    atomicAdd(&counts[idx], 1.0f);
    atomicAdd(&sLoss[0], recon);
    atomicAdd(&sLoss[1], commit);
    __syncthreads();
    if (tid == 0) {
        atomicAdd(&lossAcc[0], sLoss[0]);
        atomicAdd(&lossAcc[1], sLoss[1]);
    }
}

// ---------------------------------------------------------------------------
// Finalize: EMA, argmax, sequential dead-code reinit scan, outputs.
// out = [loss(1), new_vectors(1024*32), assign_ema(1024)]
// ---------------------------------------------------------------------------
__global__ __launch_bounds__(1024) void vq_finalize(
    const float* __restrict__ vectors, const float* __restrict__ assignments,
    const float* __restrict__ noise, const float* __restrict__ counts,
    const float* __restrict__ sumz, const float* __restrict__ lossAcc,
    float* __restrict__ out)
{
    __shared__ float sA[NBINS];
    __shared__ int   sCopy[NBINS];    // dead flag, then copy mode 0/1/2
    __shared__ float rv[NBINS];
    __shared__ int   ri[NBINS];
    __shared__ int   sMaxIdx;

    const int t = threadIdx.x;
    const float c  = counts[t];
    const float ba = c * (1.0f / (float)NROWS);
    const float ae = GAMMA_F * assignments[t] + ONE_MG * ba;
    sA[t]    = ae;
    sCopy[t] = (ae * (float)NROWS < MIN_ASSIGN_F) ? 1 : 0;   // not_assigned (pre-scan)
    rv[t] = ba; ri[t] = t;
    __syncthreads();

    // argmax(batch_assign), first index wins ties
    for (int s = 512; s > 0; s >>= 1) {
        if (t < s) {
            if (rv[t + s] > rv[t] || (rv[t + s] == rv[t] && ri[t + s] < ri[t])) {
                rv[t] = rv[t + s]; ri[t] = ri[t + s];
            }
        }
        __syncthreads();
    }
    if (t == 0) sMaxIdx = ri[0];
    __syncthreads();
    const int maxi = sMaxIdx;

    // sequential scan i = 1023..0 (matches torch pop-from-end order)
    if (t == 0) {
        bool maxDead = (sCopy[maxi] != 0);
        for (int i = NBINS - 1; i >= 0; --i) {
            if (sCopy[i]) {
                float na = sA[maxi] * 0.5f;
                sA[i] = na; sA[maxi] = na;
                // dead i below a dead maxi sees vecs[maxi] already += noise[maxi]
                sCopy[i] = (maxDead && i < maxi) ? 2 : 1;
            }
        }
    }
    __syncthreads();

    const int cp = sCopy[t];
    for (int k = 0; k < D_Z; ++k) {
        float base = vectors[t * D_Z + k];
        if (cp) {
            base = vectors[maxi * D_Z + k] + noise[t * D_Z + k];
            if (cp == 2) base += noise[maxi * D_Z + k];
        }
        float mv = (c == 0.0f) ? base : (sumz[t * D_Z + k] / c);
        out[1 + t * D_Z + k] = GAMMA_F * base + ONE_MG * mv;
    }
    out[1 + NBINS * D_Z + t] = sA[t];

    if (t == 0) {
        float recon  = 0.5f * lossAcc[0] / (float)NROWS + 32.0f * LOG_2PI;
        float commit = lossAcc[1] / ((float)NROWS * (float)D_Z);
        out[0] = recon + BETA_F * commit;
    }
}

// ---------------------------------------------------------------------------
extern "C" void kernel_launch(void* const* d_in, const int* in_sizes, int n_in,
                              void* d_out, int out_size, void* d_ws, size_t ws_size,
                              hipStream_t stream)
{
    const float* x        = (const float*)d_in[0];
    const float* W_enc    = (const float*)d_in[1];
    const float* b_enc    = (const float*)d_in[2];
    const float* vectors  = (const float*)d_in[3];
    const float* W_dec    = (const float*)d_in[4];
    const float* b_dec    = (const float*)d_in[5];
    const float* assigns  = (const float*)d_in[6];
    const float* noise    = (const float*)d_in[7];
    float* out = (float*)d_out;

    char* ws = (char*)d_ws;
    size_t off = 0;
    auto alloc = [&](size_t bytes) -> void* {
        void* p = ws + off;
        off = (off + bytes + 255) & ~(size_t)255;
        return p;
    };
    float*    decoded = (float*)alloc((size_t)NBINS * D_X * sizeof(float));
    float*    vnorm   = (float*)alloc((size_t)NBINS * sizeof(float));
    float*    countsP = (float*)alloc((size_t)NBINS * sizeof(float));
    float*    sumz    = (float*)alloc((size_t)NBINS * D_Z * sizeof(float));
    float*    lossAcc = (float*)alloc(2 * sizeof(float));
    _Float16* vecsB   = (_Float16*)alloc((size_t)64 * 32 * 16 * sizeof(_Float16));
    _Float16* wencB   = (_Float16*)alloc((size_t)2 * 2 * 32 * 16 * sizeof(_Float16));
    (void)ws_size; (void)in_sizes; (void)n_in; (void)out_size;

    vq_prep<<<128, 256, 0, stream>>>(W_enc, vectors, W_dec, b_dec, decoded, vnorm,
                                     countsP, sumz, lossAcc, vecsB, wencB);
    vq_main<<<NROWS / 128, 128, 0, stream>>>(x, b_enc, vectors, decoded, vnorm,
                                             vecsB, wencB, countsP, sumz, lossAcc);
    vq_finalize<<<1, 1024, 0, stream>>>(vectors, assigns, noise, countsP, sumz,
                                        lossAcc, out);
}
